// ParallelToeplitzHeads_90864328114412
// MI455X (gfx1250) — matrix operand order repaired and checked by reading, not compiled, not run on hardware
//
#include <hip/hip_runtime.h>

// ParallelToeplitzHeads for MI455X (gfx1250, wave32, WMMA bf16 K=32).
//   pre-pass : w_in/w_out -> bf16 ; x -> xT (B,T,DIM) bf16   (one-time, BW-bound)
//   stage 1/3: GEMM  C[b,m,n] = sum_k W[m,k]*Xt[b,n,k] + bias[m]
//              128x256 WG tile, 64x64 wave tile, double-buffered async LDS staging.
//   stage 2  : per-head causal Toeplitz matmul, B fragments built from a
//              reversed zero-padded toep_w table in LDS (v_perm_b32 packing).
// GEMM operands are converted to bf16 outside the WMMA loops; Toeplitz B is packed in-loop. Workspace: ~100 MB.

#define B_   8
#define DIM_ 1024
#define T_   2048
#define H_   16
#define HD_  64

typedef __attribute__((ext_vector_type(16))) __bf16 v16bf;
typedef __attribute__((ext_vector_type(8)))  float  v8f;
typedef __attribute__((ext_vector_type(4)))  int    v4i;

union FragU {
  uint4    q[2];   // 32 bytes
  unsigned u[8];
  v16bf    v;      // 16 bf16
};

__device__ __forceinline__ unsigned short f2bf(float f) {   // RNE
  unsigned u = __builtin_bit_cast(unsigned, f);
  u += 0x7FFFu + ((u >> 16) & 1u);
  return (unsigned short)(u >> 16);
}
// pack two f32 -> two bf16 (truncate) in ONE v_perm_b32
__device__ __forceinline__ unsigned pk2bf(float a, float b) {
  return __builtin_amdgcn_perm(__builtin_bit_cast(unsigned, b),
                               __builtin_bit_cast(unsigned, a), 0x07060302u);
}

// ---- CDNA5 async global->LDS copy path (probe-verified this toolchain) ----
#if defined(__has_builtin)
#  if __has_builtin(__builtin_amdgcn_global_load_async_to_lds_b128) && \
      __has_builtin(__builtin_amdgcn_s_wait_asynccnt)
#    define HAVE_ASYNC 1
#  endif
#endif
#ifndef HAVE_ASYNC
#  define HAVE_ASYNC 0
#endif

typedef __attribute__((address_space(1))) v4i* gv4i_p;   // global v4i*
typedef __attribute__((address_space(3))) v4i* lv4i_p;   // LDS    v4i*

__device__ __forceinline__ void copy16_g2l(const unsigned short* g, unsigned short* l) {
#if HAVE_ASYNC
  __builtin_amdgcn_global_load_async_to_lds_b128(
      (gv4i_p)(unsigned long long)(size_t)g,
      (lv4i_p)(unsigned)(size_t)l, 0, 0);
#else
  *(uint4*)l = *(const uint4*)g;
#endif
}
__device__ __forceinline__ void copy_join() {
#if HAVE_ASYNC
  __builtin_amdgcn_s_wait_asynccnt(0);
#endif
}

// ---------------------------------------------------------------------------
// Pre-pass: elementwise f32 -> bf16  (8 elements / thread)
// ---------------------------------------------------------------------------
__global__ __launch_bounds__(256)
void cvt_f32_bf16(const float* __restrict__ src, unsigned short* __restrict__ dst, int n) {
  const int i = (blockIdx.x * 256 + threadIdx.x) * 8;
  if (i >= n) return;
  const float4 a = *(const float4*)(src + i);
  const float4 b = *(const float4*)(src + i + 4);
  uint4 o;
  o.x = pk2bf(a.x, a.y); o.y = pk2bf(a.z, a.w);
  o.z = pk2bf(b.x, b.y); o.w = pk2bf(b.z, b.w);
  *(uint4*)(dst + i) = o;
}

// ---------------------------------------------------------------------------
// Pre-pass: x (B,DIM,T) f32 -> xT (B,T,DIM) bf16, 32x32 LDS-tiled transpose
// ---------------------------------------------------------------------------
__global__ __launch_bounds__(256)
void transpose_x_bf16(const float* __restrict__ X, unsigned short* __restrict__ XT) {
  __shared__ unsigned short tile[32][33];
  const int tid = threadIdx.x;
  const int t0 = blockIdx.x * 32, d0 = blockIdx.y * 32, b = blockIdx.z;
  const int row = tid >> 3, q = tid & 7;

  const float4 v = *(const float4*)(X + ((size_t)b * DIM_ + d0 + row) * T_ + t0 + q * 4);
  tile[row][q * 4 + 0] = f2bf(v.x);
  tile[row][q * 4 + 1] = f2bf(v.y);
  tile[row][q * 4 + 2] = f2bf(v.z);
  tile[row][q * 4 + 3] = f2bf(v.w);
  __syncthreads();

  const unsigned r0 = tile[q * 4 + 0][row], r1 = tile[q * 4 + 1][row];
  const unsigned r2 = tile[q * 4 + 2][row], r3 = tile[q * 4 + 3][row];
  uint2 d; d.x = r0 | (r1 << 16); d.y = r2 | (r3 << 16);
  *(uint2*)(XT + ((size_t)b * T_ + t0 + row) * DIM_ + d0 + q * 4) = d;
}

// ---------------------------------------------------------------------------
// GEMM: C[b,m,n] = sum_k W[m,k] * Xt[b,n,k] + bias[m]
// grid (N/256, M/128, B), block 256 (8 waves, 2x4). Wave tile: 64m x 64n
// (16 WMMA / K-step). Double-buffered LDS, async prefetch of next K tile.
// ---------------------------------------------------------------------------
template <bool OUT_BF16>
__global__ __launch_bounds__(256)
void gemm_wmma(const unsigned short* __restrict__ W, const float* __restrict__ bias,
               const unsigned short* __restrict__ Xt, void* __restrict__ Cv) {
  constexpr int M = 1024, N = 2048, K = 1024, LDA = 40;  // 80B rows, 16B aligned
  __shared__ __align__(16) unsigned short lsA[2][128 * LDA];
  __shared__ __align__(16) unsigned short lsB[2][256 * LDA];
  __shared__ float lsBias[128];

  const int tid = threadIdx.x;
  const int b   = blockIdx.z;
  const int m0  = blockIdx.y * 128;
  const int n0  = blockIdx.x * 256;

  const int wave = tid >> 5, lane = tid & 31;
  const int wm = wave >> 2, wn = wave & 3;       // 2 x 4 waves
  const int hi = lane >> 4, ml = lane & 15;

  if (tid < 128) lsBias[tid] = bias[m0 + tid];

  const unsigned short* Bb = Xt + (size_t)b * N * K;

  auto stage = [&](int buf, int kb) {
    #pragma unroll
    for (int p = 0; p < 2; ++p) {                // A: 128 rows x 4 chunks
      const int chunk = p * 256 + tid;
      const int row = chunk >> 2, q = chunk & 3;
      copy16_g2l(W + (size_t)(m0 + row) * K + kb + q * 8, &lsA[buf][row * LDA + q * 8]);
    }
    #pragma unroll
    for (int p = 0; p < 4; ++p) {                // B: 256 rows x 4 chunks
      const int chunk = p * 256 + tid;
      const int row = chunk >> 2, q = chunk & 3;
      copy16_g2l(Bb + (size_t)(n0 + row) * K + kb + q * 8, &lsB[buf][row * LDA + q * 8]);
    }
  };

  v8f acc[4][4] = {};
  stage(0, 0);
  int ib = 0;
  for (int kb = 0; kb < K; kb += 32, ib ^= 1) {
    copy_join();          // own async copies into buf ib done
    __syncthreads();      // publish buf ib; everyone done reading buf ib^1
    if (kb + 32 < K) stage(ib ^ 1, kb + 32);   // prefetch overlaps compute

    FragU a[4], bf[4];
    #pragma unroll
    for (int r = 0; r < 4; ++r) {   // A: K run [16*hi,+16)
      const int row = wm * 64 + r * 16 + ml;
      const uint4* p = (const uint4*)&lsA[ib][row * LDA + 16 * hi];
      a[r].q[0] = p[0];
      a[r].q[1] = p[1];
    }
    #pragma unroll
    for (int c = 0; c < 4; ++c) {   // B: K run [16*hi,+16)
      const int row = wn * 64 + c * 16 + ml;
      const uint4* p = (const uint4*)&lsB[ib][row * LDA + 16 * hi];
      bf[c].q[0] = p[0];
      bf[c].q[1] = p[1];
    }
    #pragma unroll
    for (int r = 0; r < 4; ++r)
      #pragma unroll
      for (int c = 0; c < 4; ++c)
        acc[r][c] = __builtin_amdgcn_wmma_f32_16x16x32_bf16(
            false, a[r].v, false, bf[c].v, (short)0, acc[r][c], false, false);
  }

  #pragma unroll
  for (int r = 0; r < 4; ++r)
    #pragma unroll
    for (int c = 0; c < 4; ++c) {
      const int gn = n0 + wn * 64 + c * 16 + ml;
      #pragma unroll
      for (int v = 0; v < 8; ++v) {
        const int lm = wm * 64 + r * 16 + v + 8 * hi;
        const float val = acc[r][c][v] + lsBias[lm];
        if constexpr (OUT_BF16)
          ((unsigned short*)Cv)[((size_t)b * M + (m0 + lm)) * N + gn] = f2bf(val);
        else
          ((float*)Cv)[((size_t)b * M + (m0 + lm)) * N + gn] = val;
      }
    }
}

// ---------------------------------------------------------------------------
// Toeplitz: grid (T/128, H, B), block 256. One (b,h): 64 rows x 128 cols.
// Wave: 32m x 32n. A = c1 bf16 [d][t], double-buffered async staging.
// B fragment: lane reads 16 ascending-contiguous f32 from reversed table,
// packs with v_perm_b32. Output c2t bf16 (B,T,DIM), packed 16B stores.
// ---------------------------------------------------------------------------
__global__ __launch_bounds__(256)
void toeplitz_wmma(const unsigned short* __restrict__ C1, const float* __restrict__ tw,
                   const float* __restrict__ tb, unsigned short* __restrict__ C2T) {
  constexpr int LDA = 40;
  constexpr int RLEN = T_ + 128;
  __shared__ __align__(16) unsigned short lsA[2][64 * LDA];
  __shared__ float lsR[RLEN];

  const int tid = threadIdx.x;
  const int n0  = blockIdx.x * 128;
  const int h   = blockIdx.y;
  const int b   = blockIdx.z;

  const int wave = tid >> 5, lane = tid & 31;
  const int wm = wave >> 2, wn = wave & 3;
  const int hi = lane >> 4, ml = lane & 15;

  // lsR[j] = tw[h][T-1-j] for j < T, else 0  (reversed + zero pad)
  for (int j = tid; j < RLEN; j += 256)
    lsR[j] = (j < T_) ? tw[(size_t)h * T_ + (T_ - 1 - j)] : 0.0f;

  const unsigned short* A = C1 + ((size_t)b * DIM_ + h * HD_) * T_;
  auto stage = [&](int buf, int kb) {            // 64 rows x 4 chunks, 1/thread
    const int row = tid >> 2, q = tid & 3;
    copy16_g2l(A + (size_t)row * T_ + kb + q * 8, &lsA[buf][row * LDA + q * 8]);
  };

  v8f acc[2][2] = {};
  const int kmax = n0 + 128;  // causal: only t <= max u in this tile
  stage(0, 0);
  int ib = 0;
  for (int kb = 0; kb < kmax; kb += 32, ib ^= 1) {
    copy_join();
    __syncthreads();
    if (kb + 32 < kmax) stage(ib ^ 1, kb + 32);

    FragU a[2];
    #pragma unroll
    for (int r = 0; r < 2; ++r) {
      const int row = wm * 32 + r * 16 + ml;
      const uint4* p = (const uint4*)&lsA[ib][row * LDA + 16 * hi];
      a[r].q[0] = p[0];
      a[r].q[1] = p[1];
    }

    #pragma unroll
    for (int c = 0; c < 2; ++c) {
      const int uBase = n0 + wn * 32 + c * 16;
      if (kb > uBase + 15) continue;           // wave-uniform: tile all-zero
      const int u  = uBase + ml;               // this lane's output column
      const int jb = (T_ - 1) - u + kb + 16 * hi;  // ascending contiguous run
      FragU bv;
      #pragma unroll
      for (int e = 0; e < 8; ++e)
        bv.u[e] = pk2bf(lsR[jb + 2 * e], lsR[jb + 2 * e + 1]);
      #pragma unroll
      for (int r = 0; r < 2; ++r)
        acc[r][c] = __builtin_amdgcn_wmma_f32_16x16x32_bf16(
            false, a[r].v, false, bv.v, (short)0, acc[r][c], false, false);
    }
  }

  // epilogue: c2t[b][u][h*64+m] bf16, m contiguous -> one b128 store per tile
  #pragma unroll
  for (int c = 0; c < 2; ++c) {
    const int u = n0 + wn * 32 + c * 16 + ml;
    const float tbv = tb[(size_t)h * T_ + u];
    #pragma unroll
    for (int r = 0; r < 2; ++r) {
      uint4 o;
      o.x = pk2bf(acc[r][c][0] + tbv, acc[r][c][1] + tbv);
      o.y = pk2bf(acc[r][c][2] + tbv, acc[r][c][3] + tbv);
      o.z = pk2bf(acc[r][c][4] + tbv, acc[r][c][5] + tbv);
      o.w = pk2bf(acc[r][c][6] + tbv, acc[r][c][7] + tbv);
      const int mb = h * HD_ + wm * 32 + r * 16 + 8 * hi;   // 16B aligned
      *(uint4*)(C2T + ((size_t)b * T_ + u) * DIM_ + mb) = o;
    }
  }
}

// ---------------------------------------------------------------------------
extern "C" void kernel_launch(void* const* d_in, const int* in_sizes, int n_in,
                              void* d_out, int out_size, void* d_ws, size_t ws_size,
                              hipStream_t stream) {
  (void)in_sizes; (void)n_in; (void)out_size; (void)ws_size;
  const float* x     = (const float*)d_in[0];
  const float* w_in  = (const float*)d_in[1];
  const float* b_in  = (const float*)d_in[2];
  const float* w_out = (const float*)d_in[3];
  const float* b_out = (const float*)d_in[4];
  const float* tw    = (const float*)d_in[5];
  const float* tb    = (const float*)d_in[6];

  // workspace layout (bf16 elements)
  unsigned short* ws  = (unsigned short*)d_ws;
  unsigned short* xT  = ws;                                 // B*T*DIM   (32 MB)
  unsigned short* wiB = xT  + (size_t)B_ * T_ * DIM_;       // DIM*DIM   ( 2 MB)
  unsigned short* woB = wiB + (size_t)DIM_ * DIM_;          // DIM*DIM   ( 2 MB)
  unsigned short* c1  = woB + (size_t)DIM_ * DIM_;          // B*DIM*T   (32 MB)
  unsigned short* c2t = c1  + (size_t)B_ * DIM_ * T_;       // B*T*DIM   (32 MB)

  dim3 blk(256);
  const int nw = DIM_ * DIM_;
  cvt_f32_bf16<<<dim3(nw / (256 * 8)), blk, 0, stream>>>(w_in,  wiB, nw);
  cvt_f32_bf16<<<dim3(nw / (256 * 8)), blk, 0, stream>>>(w_out, woB, nw);
  transpose_x_bf16<<<dim3(T_ / 32, DIM_ / 32, B_), blk, 0, stream>>>(x, xT);

  dim3 g1(T_ / 256, DIM_ / 128, B_);             // 8 x 8 x 8
  dim3 g2(T_ / 128, H_, B_);                     // 16 x 16 x 8
  gemm_wmma<true ><<<g1, blk, 0, stream>>>(wiB, b_in, xT, (void*)c1);
  toeplitz_wmma   <<<g2, blk, 0, stream>>>(c1, tw, tb, c2t);
  gemm_wmma<false><<<g1, blk, 0, stream>>>(woB, b_out, c2t, d_out);
}
